// Qwen3MoeAttention_11467562680416
// MI455X (gfx1250) — matrix operand order repaired and checked by reading, not compiled, not run on hardware
//
#include <hip/hip_runtime.h>
#include <hip/hip_bf16.h>

typedef __attribute__((ext_vector_type(16))) __bf16 v16bf;
typedef __attribute__((ext_vector_type(8)))  float  v8f;

#define S_LEN 2048
#define NH    32
#define NKV   4
#define HD    128
#define LOG_ROPE_THETA 13.815510557964274f   // ln(1e6)

// ---------------------------------------------------------------------------
// WMMA fragment helpers (CDNA5 16x16x32 bf16 layouts, wave32).
// A (16x32): lane<16 row=l: VGPR0-3 hold K=kg*8+0..7, VGPR4-7 K=16+kg*8+0..7
//            -> two contiguous 16B runs.
// B (32x16): lane holds column n: K = kg*8+0..7, 16+kg*8+0..7 (same as A).
// C (16x16): VGPR j: lanes<16 -> row j, lanes>=16 -> row j+8; col = lane&15.
// All rows (LDS and global) used for fragments are 16B aligned so these lower
// to ds_load_b128 / global_load_b128.
// ---------------------------------------------------------------------------
union F16x16 { uint4 u[2]; v16bf v; };

__device__ __forceinline__ v16bf frag_a_load(const __bf16* rowp, int kbase, int kg) {
  F16x16 f;
  f.u[0] = *(const uint4*)(rowp + kbase + kg * 8);
  f.u[1] = *(const uint4*)(rowp + kbase + 16 + kg * 8);
  return f.v;
}

__device__ __forceinline__ v16bf frag_b_load(const __bf16* rowp, int kbase, int kg) {
  F16x16 f;
  f.u[0] = *(const uint4*)(rowp + kbase + kg * 8);
  f.u[1] = *(const uint4*)(rowp + kbase + 16 + kg * 8);
  return f.v;
}

__device__ __forceinline__ v8f wmma_bf16(v16bf a, v16bf b, v8f c) {
  return __builtin_amdgcn_wmma_f32_16x16x32_bf16(false, a, false, b, (short)0, c,
                                                 false, false);
}

__device__ __forceinline__ v8f v8f_zero() {
  v8f z;
#pragma unroll
  for (int i = 0; i < 8; ++i) z[i] = 0.0f;
  return z;
}

// ---------------------------------------------------------------------------
// fp32 -> bf16 conversion, float4-vectorized (n divisible by 4)
// ---------------------------------------------------------------------------
__global__ __launch_bounds__(256)
void f32_to_bf16_kernel(const float* __restrict__ in, __bf16* __restrict__ out,
                        int n4) {
  for (int i = blockIdx.x * blockDim.x + threadIdx.x; i < n4;
       i += gridDim.x * blockDim.x) {
    float4 v = ((const float4*)in)[i];
    union { uint2 u; __bf16 h[4]; } o;
    o.h[0] = (__bf16)v.x; o.h[1] = (__bf16)v.y;
    o.h[2] = (__bf16)v.z; o.h[3] = (__bf16)v.w;
    ((uint2*)out)[i] = o.u;
  }
}

// ---------------------------------------------------------------------------
// fp32 W[K][N] -> bf16 Wt[N][K] tiled transpose-convert (32x32 tiles)
// ---------------------------------------------------------------------------
__global__ __launch_bounds__(256)
void transpose_convert_kernel(const float* __restrict__ in, __bf16* __restrict__ out,
                              int K, int N) {
  __shared__ float T[32][33];
  const int k0 = blockIdx.y * 32, n0 = blockIdx.x * 32;
  const int tx = threadIdx.x & 31, ty = threadIdx.x >> 5;   // ty: 0..7
#pragma unroll
  for (int i = 0; i < 32; i += 8)
    T[ty + i][tx] = in[(size_t)(k0 + ty + i) * N + (n0 + tx)];
  __syncthreads();
#pragma unroll
  for (int i = 0; i < 32; i += 8)
    out[(size_t)(n0 + ty + i) * K + (k0 + tx)] = (__bf16)T[tx][ty + i];
}

// ---------------------------------------------------------------------------
// Tiled bf16 GEMM  C[M,N] = A[M,K] * Bt[N,K]^T   (A row-major, Bt transposed)
// Block tile 128x128, KT=32, 256 threads = 8 waves (4 M x 2 N); wave tile
// 32x64 -> 8 WMMAs per 12 ds_load_b128. Double-buffered LDS with register-
// staged global prefetch. The fp32 epilogue staging tile overlays the (dead)
// double buffers; norm/store epilogues run in two 64-row passes.
// MODE 0: Q epilogue (RMSNorm+RoPE -> bf16 [B,NH,S,HD])
// MODE 1: K epilogue (RMSNorm+RoPE -> bf16 [B,NKV,S,HD])
// MODE 2: V epilogue (-> bf16 [B,NKV,HD,S], d-major for the attention kernel)
// MODE 3: OUT epilogue (-> fp32 row-major [M,N], vectorized through LDS)
// ---------------------------------------------------------------------------
enum { MODE_Q = 0, MODE_K = 1, MODE_V = 2, MODE_OUT = 3 };

template <int MODE>
__global__ __launch_bounds__(256)
void gemm_proj_kernel(const __bf16* __restrict__ A, const __bf16* __restrict__ Bt,
                      void* __restrict__ Cout, int M, int N, int K,
                      const float* __restrict__ normw, int nheads) {
  __shared__ __align__(16) char smem[40960];
  __bf16 (*AsB)[40] = (__bf16(*)[40])smem;            // [2*128][40]
  __bf16 (*BsB)[40] = (__bf16(*)[40])(smem + 20480);  // [2*128][40]
  float  (*Cst)[128] = (float(*)[128])smem;           // [64][128] overlay

  const int tid  = threadIdx.x;
  const int lane = tid & 31;
  const int wid  = tid >> 5;
  const int wm   = wid >> 1;          // 0..3 : rows wm*32
  const int wn   = wid & 1;           // 0..1 : cols wn*64
  const int kg   = lane >> 4;
  const int ln   = lane & 15;
  const int m0   = blockIdx.y * 128;
  const int n0   = blockIdx.x * 128;

  // staging: two uint4 (8 halfs) per thread for A and for B (128 x 32 each)
  const int sr = tid >> 2, sc = (tid & 3) * 8;

  v8f acc0[4], acc1[4];
#pragma unroll
  for (int t = 0; t < 4; ++t) { acc0[t] = v8f_zero(); acc1[t] = v8f_zero(); }

  const int NT = K >> 5;

  uint4 a0 = *(const uint4*)&A [(size_t)(m0 + sr) * K + sc];
  uint4 a1 = *(const uint4*)&A [(size_t)(m0 + sr + 64) * K + sc];
  uint4 b0 = *(const uint4*)&Bt[(size_t)(n0 + sr) * K + sc];
  uint4 b1 = *(const uint4*)&Bt[(size_t)(n0 + sr + 64) * K + sc];
  *(uint4*)&AsB[sr][sc]      = a0;
  *(uint4*)&AsB[sr + 64][sc] = a1;
  *(uint4*)&BsB[sr][sc]      = b0;
  *(uint4*)&BsB[sr + 64][sc] = b1;

  for (int kt = 0; kt < NT; ++kt) {
    __syncthreads();
    const int cur = (kt & 1) * 128;
    const bool more = (kt + 1) < NT;
    if (more) {
      const int k0 = (kt + 1) << 5;
      a0 = *(const uint4*)&A [(size_t)(m0 + sr) * K + k0 + sc];
      a1 = *(const uint4*)&A [(size_t)(m0 + sr + 64) * K + k0 + sc];
      b0 = *(const uint4*)&Bt[(size_t)(n0 + sr) * K + k0 + sc];
      b1 = *(const uint4*)&Bt[(size_t)(n0 + sr + 64) * K + k0 + sc];
    }
    v16bf af0 = frag_a_load(&AsB[cur + wm * 32 + ln][0], 0, kg);
    v16bf af1 = frag_a_load(&AsB[cur + wm * 32 + 16 + ln][0], 0, kg);
#pragma unroll
    for (int t = 0; t < 4; ++t) {
      v16bf bf = frag_b_load(&BsB[cur + wn * 64 + t * 16 + ln][0], 0, kg);
      acc0[t] = wmma_bf16(af0, bf, acc0[t]);
      acc1[t] = wmma_bf16(af1, bf, acc1[t]);
    }
    if (more) {
      const int nxt = (cur ^ 128);
      *(uint4*)&AsB[nxt + sr][sc]      = a0;
      *(uint4*)&AsB[nxt + sr + 64][sc] = a1;
      *(uint4*)&BsB[nxt + sr][sc]      = b0;
      *(uint4*)&BsB[nxt + sr + 64][sc] = b1;
    }
  }

  if (MODE == MODE_V) {
    // store bf16 [B,NKV,HD,S]: per lane, 8 consecutive s values -> one b128
    __bf16* C = (__bf16*)Cout;
#pragma unroll
    for (int mi = 0; mi < 2; ++mi)
#pragma unroll
      for (int t = 0; t < 4; ++t) {
        int col = n0 + wn * 64 + t * 16 + ln;
        int h = col >> 7, d = col & (HD - 1);
        int s0 = m0 + wm * 32 + mi * 16 + ((lane < 16) ? 0 : 8);
        int b  = s0 >> 11;
        int s  = s0 & (S_LEN - 1);
        union { uint4 u; __bf16 h8[8]; } st;
        const v8f& ac = mi ? acc1[t] : acc0[t];
#pragma unroll
        for (int j = 0; j < 8; ++j) st.h8[j] = (__bf16)ac[j];
        *(uint4*)&C[((size_t)(b * nheads + h) * HD + d) * S_LEN + s] = st.u;
      }
  } else {
    // two 64-row passes through the fp32 staging tile
    for (int p = 0; p < 2; ++p) {
      __syncthreads();
      if ((wm >> 1) == p) {
        const int rbase = (wm & 1) * 32;
#pragma unroll
        for (int mi = 0; mi < 2; ++mi)
#pragma unroll
          for (int t = 0; t < 4; ++t) {
            const v8f& ac = mi ? acc1[t] : acc0[t];
#pragma unroll
            for (int j = 0; j < 8; ++j) {
              int row = rbase + mi * 16 + ((lane < 16) ? j : j + 8);
              Cst[row][wn * 64 + t * 16 + ln] = ac[j];
            }
          }
      }
      __syncthreads();

      if (MODE == MODE_OUT) {
        float* C = (float*)Cout;
#pragma unroll
        for (int t = 0; t < 8; ++t) {
          int idx = tid + t * 256;               // 2048 float4 chunks
          int r = idx >> 5, c4 = (idx & 31) * 4;
          *(float4*)&C[(size_t)(m0 + p * 64 + r) * N + n0 + c4] =
              *(float4*)&Cst[r][c4];
        }
      } else {
        // Q / K: fused RMSNorm + RoPE (BN == HD: block owns full head rows)
        __bf16* C = (__bf16*)Cout;
        const int row = tid >> 2;     // 4 threads per row, 32 elements each
        const int seg = tid & 3;
        float ss = 0.0f;
#pragma unroll
        for (int i = 0; i < 32; ++i) {
          float v = Cst[row][seg * 32 + i];
          ss += v * v;
        }
        ss += __shfl_xor(ss, 1);
        ss += __shfl_xor(ss, 2);
        const float scale = rsqrtf(ss * (1.0f / 128.0f) + 1e-6f);

        const int rowg = m0 + p * 64 + row;
        const int b = rowg >> 11, s = rowg & (S_LEN - 1);
        const int h = n0 >> 7;
        const size_t base = ((size_t)(b * nheads + h) * S_LEN + s) * HD;
#pragma unroll
        for (int c = 0; c < 4; ++c) {
          union { uint4 u; __bf16 h8[8]; } st;
#pragma unroll
          for (int i = 0; i < 8; ++i) {
            int d  = seg * 32 + c * 8 + i;
            int pd = (d < 64) ? d + 64 : d - 64;
            float nv  = Cst[row][d]  * scale * normw[d];
            float pv  = Cst[row][pd] * scale * normw[pd];
            float rot = (d < 64) ? -pv : pv;
            float ang = (float)s * __expf(-(float)(d & 63) * (LOG_ROPE_THETA / 64.0f));
            st.h8[i] = (__bf16)(nv * __cosf(ang) + rot * __sinf(ang));
          }
          *(uint4*)&C[base + seg * 32 + c * 8] = st.u;
        }
      }
    }
  }
}

// ---------------------------------------------------------------------------
// Causal GQA flash attention.
// Grid: (S/128, NH, B). 256 threads = 8 waves; wave w owns query rows
// [w*16, w*16+16). Q is loop-invariant per wave, so its 4 A-fragments are
// loaded ONCE directly from global (16B runs) -> no Q LDS tile, no barrier.
// KV tiles of 64 rows; V comes d-major so both staging sides are 128-bit.
// Online softmax: rows 0-7 live in lanes 0-15, rows 8-15 in lanes 16-31, so
// shfl_xor {1,2,4,8} reduces exactly one row. The O write-out buffer overlays
// the K/V/P tiles after the loop.
// ---------------------------------------------------------------------------
__global__ __launch_bounds__(256)
void flash_attn_kernel(const __bf16* __restrict__ Qg, const __bf16* __restrict__ Kg,
                       const __bf16* __restrict__ Vg, __bf16* __restrict__ attn) {
  __shared__ __align__(16) char smem[54272];
  __bf16 (*Ks)[136] = (__bf16(*)[136])smem;            // [64][136]  [kvrow][d]
  __bf16 (*Vs)[72]  = (__bf16(*)[72])(smem + 17408);   // [128][72]  [d][kvrow]
  __bf16 (*Ps)[72]  = (__bf16(*)[72])(smem + 35840);   // [128][72]  [q][kvrow]
  __bf16 (*Os)[136] = (__bf16(*)[136])smem;            // [128][136] overlay

  const int tid  = threadIdx.x;
  const int lane = tid & 31;
  const int wid  = tid >> 5;
  const int kg   = lane >> 4;
  const int ln   = lane & 15;
  const int q0   = blockIdx.x * 128;
  const int h    = blockIdx.y;
  const int bb   = blockIdx.z;
  const int kvh  = h >> 3;                 // NH/NKV = 8
  const int m0w  = wid * 16;

  const size_t qbase = (size_t)(bb * NH + h) * S_LEN * HD;
  const size_t kbase = (size_t)(bb * NKV + kvh) * S_LEN * HD;
  const size_t vbase = (size_t)(bb * NKV + kvh) * HD * S_LEN;  // d-major

  // Q fragments straight from global memory (held in VGPRs for the whole loop)
  const __bf16* qrow = Qg + qbase + (size_t)(q0 + m0w + ln) * HD;
  v16bf qa[4];
#pragma unroll
  for (int ks = 0; ks < 4; ++ks) qa[ks] = frag_a_load(qrow, ks * 32, kg);

  v8f o[8];
#pragma unroll
  for (int t = 0; t < 8; ++t) o[t] = v8f_zero();
  float m_r[8], l_r[8];
#pragma unroll
  for (int j = 0; j < 8; ++j) { m_r[j] = -3.0e38f; l_r[j] = 0.0f; }

  const float scaling = 0.08838834764831845f;  // HD^-0.5

  for (int j0 = 0; j0 < q0 + 128; j0 += 64) {
    __syncthreads();
#pragma unroll
    for (int t = 0; t < 4; ++t) {          // K tile: 1024 uint4 chunks
      int idx = tid + t * 256;
      int r = idx >> 4, c8 = (idx & 15) * 8;
      *(uint4*)&Ks[r][c8] = *(const uint4*)&Kg[kbase + (size_t)(j0 + r) * HD + c8];
    }
#pragma unroll
    for (int t = 0; t < 4; ++t) {          // V tile (d-major): 1024 uint4
      int idx = tid + t * 256;
      int d = idx >> 3, r8 = (idx & 7) * 8;
      *(uint4*)&Vs[d][r8] = *(const uint4*)&Vg[vbase + (size_t)d * S_LEN + j0 + r8];
    }
    __syncthreads();

    // S = Q * K^T  (16 q rows x 64 kv cols per wave)
    v8f sacc[4];
#pragma unroll
    for (int t = 0; t < 4; ++t) sacc[t] = v8f_zero();
#pragma unroll
    for (int ks = 0; ks < 4; ++ks)
#pragma unroll
      for (int t = 0; t < 4; ++t) {
        v16bf bf = frag_b_load(&Ks[t * 16 + ln][0], ks * 32, kg);
        sacc[t] = wmma_bf16(qa[ks], bf, sacc[t]);
      }

    // scale + causal mask + online softmax
    const bool domask = (j0 >= q0);
    float mcur[8];
#pragma unroll
    for (int j = 0; j < 8; ++j) {
      int rowg = q0 + m0w + ((lane < 16) ? j : j + 8);
      float mx = -3.0e38f;
#pragma unroll
      for (int t = 0; t < 4; ++t) {
        float s = sacc[t][j] * scaling;
        if (domask && (j0 + t * 16 + ln > rowg)) s = -3.0e38f;
        sacc[t][j] = s;
        mx = fmaxf(mx, s);
      }
      mx = fmaxf(mx, __shfl_xor(mx, 1));
      mx = fmaxf(mx, __shfl_xor(mx, 2));
      mx = fmaxf(mx, __shfl_xor(mx, 4));
      mx = fmaxf(mx, __shfl_xor(mx, 8));
      mcur[j] = mx;
    }
#pragma unroll
    for (int j = 0; j < 8; ++j) {
      float mnew  = fmaxf(m_r[j], mcur[j]);
      float alpha = __expf(m_r[j] - mnew);
      float rs = 0.0f;
      int rloc = m0w + ((lane < 16) ? j : j + 8);
#pragma unroll
      for (int t = 0; t < 4; ++t) {
        float e = __expf(sacc[t][j] - mnew);
        rs += e;
        Ps[rloc][t * 16 + ln] = (__bf16)e;
      }
      rs += __shfl_xor(rs, 1);
      rs += __shfl_xor(rs, 2);
      rs += __shfl_xor(rs, 4);
      rs += __shfl_xor(rs, 8);
      l_r[j] = l_r[j] * alpha + rs;
      m_r[j] = mnew;
#pragma unroll
      for (int dt = 0; dt < 8; ++dt) o[dt][j] *= alpha;
    }
    __syncthreads();

    // O += P * V  (16 q rows x 128 d, k=64 in two 32-steps)
#pragma unroll
    for (int ks = 0; ks < 2; ++ks) {
      v16bf a = frag_a_load(&Ps[m0w + ln][0], ks * 32, kg);
#pragma unroll
      for (int dt = 0; dt < 8; ++dt) {
        v16bf bf = frag_b_load(&Vs[dt * 16 + ln][0], ks * 32, kg);
        o[dt] = wmma_bf16(a, bf, o[dt]);
      }
    }
  }

  // normalize, stage O (overlaying K/V/P), then 128-bit stores to [B*S, NH*HD]
  __syncthreads();
#pragma unroll
  for (int j = 0; j < 8; ++j) {
    float inv = 1.0f / l_r[j];
    int rloc = m0w + ((lane < 16) ? j : j + 8);
#pragma unroll
    for (int dt = 0; dt < 8; ++dt)
      Os[rloc][dt * 16 + ln] = (__bf16)(o[dt][j] * inv);
  }
  __syncthreads();
#pragma unroll
  for (int t = 0; t < 8; ++t) {
    int idx = tid + t * 256;
    int r = idx >> 4, c8 = (idx & 15) * 8;
    size_t dst = ((size_t)bb * S_LEN + q0 + r) * (NH * HD) + (size_t)h * HD + c8;
    *(uint4*)&attn[dst] = *(const uint4*)&Os[r][c8];
  }
}

// ---------------------------------------------------------------------------
// Host-side launch
// ---------------------------------------------------------------------------
extern "C" void kernel_launch(void* const* d_in, const int* in_sizes, int n_in,
                              void* d_out, int out_size, void* d_ws, size_t ws_size,
                              hipStream_t stream) {
  const float* hid = (const float*)d_in[0];
  const float* Wq  = (const float*)d_in[1];
  const float* Wk  = (const float*)d_in[2];
  const float* Wv  = (const float*)d_in[3];
  const float* Wo  = (const float*)d_in[4];
  const float* qnw = (const float*)d_in[5];
  const float* knw = (const float*)d_in[6];
  float* out = (float*)d_out;

  const int M    = 2 * S_LEN;      // 4096 token rows
  const int HID  = 2048;
  const int NQ   = NH * HD;        // 4096
  const int NKVC = NKV * HD;       // 512

  char* ws = (char*)d_ws;
  size_t off = 0;
  auto alloc = [&](size_t bytes) {
    void* p = ws + off;
    off = (off + bytes + 255) & ~(size_t)255;
    return p;
  };

  __bf16* hb   = (__bf16*)alloc((size_t)M * HID * 2);    // hidden bf16 [M][HID]
  __bf16* wqt  = (__bf16*)alloc((size_t)NQ * HID * 2);   // Wq^T  [NQ][HID]
  __bf16* wkt  = (__bf16*)alloc((size_t)NKVC * HID * 2); // Wk^T  [NKVC][HID]
  __bf16* wvt  = (__bf16*)alloc((size_t)NKVC * HID * 2); // Wv^T  [NKVC][HID]
  __bf16* wot  = (__bf16*)alloc((size_t)HID * NQ * 2);   // Wo^T  [HID][NQ]
  __bf16* Qt   = (__bf16*)alloc((size_t)M * NQ * 2);     // [B,NH,S,HD]
  __bf16* Kt   = (__bf16*)alloc((size_t)M * NKVC * 2);   // [B,NKV,S,HD]
  __bf16* Vt   = (__bf16*)alloc((size_t)M * NKVC * 2);   // [B,NKV,HD,S]
  __bf16* attn = (__bf16*)alloc((size_t)M * NQ * 2);     // [B*S, NH*HD]

  f32_to_bf16_kernel<<<1024, 256, 0, stream>>>(hid, hb, (M * HID) / 4);
  transpose_convert_kernel<<<dim3(NQ / 32,  HID / 32), 256, 0, stream>>>(Wq, wqt, HID, NQ);
  transpose_convert_kernel<<<dim3(NKVC / 32, HID / 32), 256, 0, stream>>>(Wk, wkt, HID, NKVC);
  transpose_convert_kernel<<<dim3(NKVC / 32, HID / 32), 256, 0, stream>>>(Wv, wvt, HID, NKVC);
  transpose_convert_kernel<<<dim3(HID / 32,  NQ / 32), 256, 0, stream>>>(Wo, wot, NQ, HID);

  gemm_proj_kernel<MODE_Q><<<dim3(NQ / 128, M / 128), 256, 0, stream>>>(
      hb, wqt, (void*)Qt, M, NQ, HID, qnw, NH);
  gemm_proj_kernel<MODE_K><<<dim3(NKVC / 128, M / 128), 256, 0, stream>>>(
      hb, wkt, (void*)Kt, M, NKVC, HID, knw, NKV);
  gemm_proj_kernel<MODE_V><<<dim3(NKVC / 128, M / 128), 256, 0, stream>>>(
      hb, wvt, (void*)Vt, M, NKVC, HID, nullptr, NKV);

  flash_attn_kernel<<<dim3(S_LEN / 128, NH, 2), 256, 0, stream>>>(Qt, Kt, Vt, attn);

  gemm_proj_kernel<MODE_OUT><<<dim3(HID / 128, M / 128), 256, 0, stream>>>(
      attn, wot, (void*)out, M, HID, NQ, nullptr, 0);
}
